// AttentionHead_50139448214194
// MI455X (gfx1250) — compile-verified, matrix operand order checked by reading, not run on hardware
//
#include <hip/hip_runtime.h>
#include <hip/hip_bf16.h>

// Problem constants (match reference)
#define B_   4
#define S_   4096
#define DIN_ 1024
#define D_   64      // DQK == DV

typedef __attribute__((ext_vector_type(16))) _Float16 v16h;
typedef __attribute__((ext_vector_type(8)))  _Float16 v8h;
typedef __attribute__((ext_vector_type(8)))  float    v8f;
typedef __attribute__((ext_vector_type(4)))  float    v4f;
typedef __attribute__((ext_vector_type(4)))  int      v4i;

static __device__ inline v16h pack16(v8h lo, v8h hi) {
  return __builtin_shufflevector(lo, hi, 0,1,2,3,4,5,6,7,8,9,10,11,12,13,14,15);
}

static __device__ inline v8f wmma_f16(v16h a, v16h b, v8f c) {
  // D(f32 16x16) = A(f16 16x32) * B(f16 32x16) + C
  return __builtin_amdgcn_wmma_f32_16x16x32_f16(false, a, false, b, (short)0, c,
                                                false, false);
}

// ---------------------------------------------------------------------------
// Kernel 0: transpose + convert weights to f16:  Wt[w][e][d] = W_w[d][e]
// ---------------------------------------------------------------------------
__global__ void __launch_bounds__(256)
wt_kernel(const float* __restrict__ Wq, const float* __restrict__ Wk,
          const float* __restrict__ Wv, _Float16* __restrict__ wt) {
  const int idx = blockIdx.x * blockDim.x + threadIdx.x;  // 0 .. 3*64*1024-1
  const int w   = idx >> 16;
  const int rem = idx & 65535;
  const int e   = rem >> 10;
  const int d   = rem & 1023;
  const float* W = (w == 0) ? Wq : (w == 1) ? Wk : Wv;
  wt[(size_t)idx] = (_Float16)W[d * D_ + e];
}

// ---------------------------------------------------------------------------
// Kernel 1 (x3 instantiations): projections. One wave -> one 16x64 tile.
// MODE 0: Q (pre-scaled by 1/sqrt(DQK+1e-8), row-major f16)
// MODE 1: K (row-major f16)
// MODE 2: V (stored transposed Vt[b][dv][s], packed b128 stores)
// ---------------------------------------------------------------------------
template <int MODE>
__global__ void __launch_bounds__(128)
proj_kernel(const float* __restrict__ in, const float* __restrict__ bias,
            const _Float16* __restrict__ W,        // [64][1024] transposed f16
            _Float16* __restrict__ outp) {
  const int lane = threadIdx.x & 31;
  const int h = lane >> 4;            // lane half (K split in A/B layouts)
  const int n = lane & 15;            // row id (A) / col id (B, C/D)
  const int w = threadIdx.x >> 5;
  const int gw   = blockIdx.x * 4 + w;       // 0..1023
  const int b    = gw >> 8;
  const int row0 = (gw & 255) << 4;

  const float* arow = in + ((size_t)b * S_ + row0 + n) * DIN_;

  v8f acc[4] = {};

  for (int kc = 0; kc < DIN_; kc += 32) {
    // A operand: 16 rows x 32 K of the streamed f32 input -> f16 (NT loads)
    v16h a;
    {
      const float* p0 = arow + kc + 8 * h;        // K = 8h..8h+7
      const float* p1 = arow + kc + 16 + 8 * h;   // K = 16+8h..16+8h+7
      v4f x0 = __builtin_nontemporal_load((const v4f*)p0);
      v4f x1 = __builtin_nontemporal_load((const v4f*)(p0 + 4));
      v4f y0 = __builtin_nontemporal_load((const v4f*)p1);
      v4f y1 = __builtin_nontemporal_load((const v4f*)(p1 + 4));
#pragma unroll
      for (int i = 0; i < 4; ++i) {
        a[i]      = (_Float16)x0[i];
        a[4 + i]  = (_Float16)x1[i];
        a[8 + i]  = (_Float16)y0[i];
        a[12 + i] = (_Float16)y1[i];
      }
    }
#pragma unroll
    for (int t = 0; t < 4; ++t) {
      const _Float16* wrow = W + (size_t)(t * 16 + n) * DIN_ + kc;
      v16h bt = pack16(*(const v8h*)(wrow + 8 * h),
                       *(const v8h*)(wrow + 16 + 8 * h));
      acc[t] = wmma_f16(a, bt, acc[t]);
    }
  }

  const float osc = (MODE == 0) ? (1.0f / sqrtf((float)D_ + 1e-8f)) : 1.0f;

  if constexpr (MODE == 2) {
    // Transposed store: lane owns 8 consecutive rows of column `col`.
#pragma unroll
    for (int t = 0; t < 4; ++t) {
      const int col = t * 16 + n;
      const float bval = bias[col];
      v8h pk;
#pragma unroll
      for (int r = 0; r < 8; ++r) pk[r] = (_Float16)(acc[t][r] + bval);
      *(v8h*)(outp + ((size_t)b * D_ + col) * S_ + row0 + 8 * h) = pk;
    }
  } else {
    // Row-major store: re-tile through LDS so global stores are b128.
    __shared__ _Float16 tbuf[4][16 * 64];
#pragma unroll
    for (int t = 0; t < 4; ++t) {
      const int col = t * 16 + n;
      const float bval = bias[col];
#pragma unroll
      for (int r = 0; r < 8; ++r)
        tbuf[w][(r + 8 * h) * 64 + col] = (_Float16)((acc[t][r] + bval) * osc);
    }
    asm volatile("s_wait_dscnt 0" ::: "memory");   // wave-synchronous LDS use
    const int row = lane >> 1, seg = lane & 1;     // 32 f16 (64B) per lane
    const _Float16* src = &tbuf[w][row * 64 + seg * 32];
    _Float16* dst = outp + ((size_t)b * S_ + row0 + row) * D_ + seg * 32;
#pragma unroll
    for (int c = 0; c < 4; ++c)
      *(v8h*)(dst + 8 * c) = *(const v8h*)(src + 8 * c);
  }
}

// ---------------------------------------------------------------------------
// Kernel 2: single-pass (flash) attention, fully transposed scheme.
//   S^T tile = K_tile(A) @ Q^T(B)  -> lane owns ONE query column, 16 key rows
//   => softmax stats are per-lane scalars (one cross-half shuffle each)
//   => P is natively in the B-operand layout for O^T = V^T(A) @ P^T(B)
//   => mask loads are contiguous b128; output stores are register-direct.
// No LDS, no per-iteration transposes.
// ---------------------------------------------------------------------------
__global__ void __launch_bounds__(128)
attn_kernel(const _Float16* __restrict__ Qh, const _Float16* __restrict__ Kh,
            const _Float16* __restrict__ Vt, const int* __restrict__ mask,
            float* __restrict__ out) {
  const int lane = threadIdx.x & 31;
  const int h = lane >> 4;
  const int n = lane & 15;
  const int gw = blockIdx.x * 4 + (threadIdx.x >> 5);
  const int b  = gw >> 8;
  const int q0 = (gw & 255) << 4;

  // Q rows as B-operand (A-style layout of Q); Q is pre-scaled by 1/sqrt(d)
  v16h qa[2];
  {
    const _Float16* qrow = Qh + ((size_t)b * S_ + q0 + n) * D_;
#pragma unroll
    for (int c = 0; c < 2; ++c)
      qa[c] = pack16(*(const v8h*)(qrow + c * 32 + 8 * h),
                     *(const v8h*)(qrow + c * 32 + 16 + 8 * h));
  }

  v8f o[4] = {};                     // O^T: lane = query col, VGPR = dv row
  float m_run = -__builtin_inff();
  float l_run = 0.0f;

  const int*      mrow  = mask + ((size_t)b * S_ + q0 + n) * S_;  // this lane's query row
  const _Float16* kbase = Kh + (size_t)b * S_ * D_;
  const _Float16* vbase = Vt + (size_t)b * D_ * S_;

  for (int k0 = 0; k0 < S_; k0 += 32) {
    // ---- S^T tiles: st[t][r] = score(query q0+n, key k0+16t+8h+r) ----
    v8f st[2] = {};
#pragma unroll
    for (int t = 0; t < 2; ++t) {
      const _Float16* krow = kbase + (size_t)(k0 + t * 16 + n) * D_;
#pragma unroll
      for (int c = 0; c < 2; ++c) {
        v16h ka = pack16(*(const v8h*)(krow + c * 32 + 8 * h),
                         *(const v8h*)(krow + c * 32 + 16 + 8 * h));
        st[t] = wmma_f16(ka, qa[c], st[t]);    // A=K tile, B=Q^T
      }
    }
    // ---- per-lane (per-query) block max over 32 keys ----
    float mb = st[0][0];
#pragma unroll
    for (int t = 0; t < 2; ++t)
#pragma unroll
      for (int r = 0; r < 8; ++r) mb = fmaxf(mb, st[t][r]);
    mb = fmaxf(mb, __shfl_xor(mb, 16, 32));    // merge the two key half-sets

    const float mnew  = fmaxf(m_run, mb);
    const float alpha = __expf(m_run - mnew);  // first block: exp(-inf)=0
    m_run = mnew;

    // ---- P = exp(S - m_new) * mask (mask is contiguous in k per lane) ----
    float pv[2][8];
#pragma unroll
    for (int t = 0; t < 2; ++t) {
      const int* mp = mrow + k0 + t * 16 + 8 * h;
      v4i mk0 = __builtin_nontemporal_load((const v4i*)mp);
      v4i mk1 = __builtin_nontemporal_load((const v4i*)(mp + 4));
#pragma unroll
      for (int r = 0; r < 4; ++r) {
        pv[t][r]     = (mk0[r] != 0) ? __expf(st[t][r]     - mnew) : 0.0f;
        pv[t][4 + r] = (mk1[r] != 0) ? __expf(st[t][4 + r] - mnew) : 0.0f;
      }
    }
    // ---- per-lane (per-query) row sum ----
    float rs = 0.0f;
#pragma unroll
    for (int t = 0; t < 2; ++t)
#pragma unroll
      for (int r = 0; r < 8; ++r) rs += pv[t][r];
    rs += __shfl_xor(rs, 16, 32);
    l_run = l_run * alpha + rs;

    // ---- P^T B-operand: pv is already laid out as A-style-of-P ----
    v16h pa;
#pragma unroll
    for (int i = 0; i < 8; ++i) {
      pa[i]     = (_Float16)pv[0][i];
      pa[8 + i] = (_Float16)pv[1][i];
    }
    // ---- rescale O^T by alpha (per-lane scalar), accumulate V^T @ P^T ----
#pragma unroll
    for (int t = 0; t < 4; ++t)
#pragma unroll
      for (int r = 0; r < 8; ++r) o[t][r] *= alpha;
#pragma unroll
    for (int t = 0; t < 4; ++t) {
      const _Float16* vrow = vbase + (size_t)(t * 16 + n) * S_ + k0;
      v16h va = pack16(*(const v8h*)(vrow + 8 * h),
                       *(const v8h*)(vrow + 16 + 8 * h));
      o[t] = wmma_f16(va, pa, o[t]);           // A=V^T tile, B=P^T
    }
  }

  // ---- epilogue: normalize, register-direct coalesced NT stores ----
  const float inv = 1.0f / l_run;              // all-masked row -> NaN (matches ref)
  float* orow = out + ((size_t)b * S_ + q0 + n) * D_;
#pragma unroll
  for (int t = 0; t < 4; ++t) {
    v4f lo, hi;
#pragma unroll
    for (int r = 0; r < 4; ++r) { lo[r] = o[t][r] * inv; hi[r] = o[t][4 + r] * inv; }
    __builtin_nontemporal_store(lo, (v4f*)(orow + t * 16 + 8 * h));
    __builtin_nontemporal_store(hi, (v4f*)(orow + t * 16 + 8 * h + 4));
  }
}

// ---------------------------------------------------------------------------
extern "C" void kernel_launch(void* const* d_in, const int* in_sizes, int n_in,
                              void* d_out, int out_size, void* d_ws, size_t ws_size,
                              hipStream_t stream) {
  const float* query = (const float*)d_in[0];
  const float* key   = (const float*)d_in[1];
  const float* value = (const float*)d_in[2];
  const int*   mask  = (const int*)  d_in[3];
  const float* Wq    = (const float*)d_in[4];
  const float* bq    = (const float*)d_in[5];
  const float* Wk    = (const float*)d_in[6];
  const float* bk    = (const float*)d_in[7];
  const float* Wv    = (const float*)d_in[8];
  const float* bv    = (const float*)d_in[9];
  float* out = (float*)d_out;

  // Workspace layout (~6.4 MB; L2-resident during attention):
  //   [0,2MB)   Qh  f16 [B][S][64]   (pre-scaled by 1/sqrt(DQK+1e-8))
  //   [2,4MB)   Kh  f16 [B][S][64]
  //   [4,6MB)   Vt  f16 [B][64][S]   (transposed V)
  //   [6MB,..)  Wt  f16 [3][64][1024] (transposed weights)
  char* ws = (char*)d_ws;
  _Float16* Qh = (_Float16*)(ws);
  _Float16* Kh = (_Float16*)(ws + (size_t)2 * 1024 * 1024);
  _Float16* Vt = (_Float16*)(ws + (size_t)4 * 1024 * 1024);
  _Float16* Wt = (_Float16*)(ws + (size_t)6 * 1024 * 1024);

  wt_kernel<<<768, 256, 0, stream>>>(Wq, Wk, Wv, Wt);
  proj_kernel<0><<<256, 128, 0, stream>>>(query, bq, Wt,             Qh);
  proj_kernel<1><<<256, 128, 0, stream>>>(key,   bk, Wt + 1 * 65536, Kh);
  proj_kernel<2><<<256, 128, 0, stream>>>(value, bv, Wt + 2 * 65536, Vt);
  attn_kernel<<<256, 128, 0, stream>>>(Qh, Kh, Vt, mask, out);
}
